// Model_5523327943274
// MI455X (gfx1250) — matrix operand order repaired and checked by reading, not compiled, not run on hardware
//
#include <hip/hip_runtime.h>
#include <hip/hip_bf16.h>

typedef __attribute__((ext_vector_type(16))) _Float16 v16h;
typedef __attribute__((ext_vector_type(8)))  _Float16 v8h;
typedef __attribute__((ext_vector_type(8)))  float    v8f;

// ---------------- model constants (fixed problem size) ----------------
constexpr int L_C  = 12;
constexpr int H_C  = 768;
constexpr int NH_C = 12;
constexpr int DH_C = 64;
constexpr int FF_C = 3072;
constexpr int T_C  = 256;
constexpr int B_C  = 32;
constexpr int KC   = 35;       // CRF states
constexpr int START_C = 33;
constexpr int END_C   = 34;
constexpr int M_C  = B_C * T_C; // 8192 rows

enum { F_GELU = 1, F_STORE_VT = 2 };

// ---------------------------------------------------------------------
// Generic batched WMMA GEMM:  C[z] = act( alpha * A[z] @ B[z]^T + bias + residual )
//   A  : M x K, row-major f16 (lda), batch offset zb*sAb + zh*sAh  (z = zb*NH + zh)
//   Bg : N x K, row-major f16 (= B^T), so each lane reads a contiguous column of B
//   Cf : optional fp32 output (ldc);  Ch : optional f16 output (ldch)
//   F_STORE_VT: store f16 output in per-head transposed layout vT[(b*NH+h)*DH+d][t]
// Block = 256 threads = 8 waves arranged 2(m) x 4(n); each wave computes a 32x32
// output tile = 2x2 WMMA tiles -> 4 x V_WMMA_F32_16X16X32_F16 per K-step with
// 2 A-fragments + 2 B-fragments (register-level reuse, ~16 FLOP/byte).
// ---------------------------------------------------------------------
__global__ __launch_bounds__(256)
void wmma_gemm_kernel(const _Float16* __restrict__ Ag, int lda, long long sAb, long long sAh,
                      const _Float16* __restrict__ Bg, int ldb, long long sBb, long long sBh,
                      const float* __restrict__ bias,
                      const float* __restrict__ residual, int ldr,
                      float* __restrict__ Cf, int ldc, long long sCb, long long sCh,
                      _Float16* __restrict__ Ch, int ldch, long long sChb, long long sChh,
                      int M, int N, int Kd, float alpha, int flags)
{
    const int z  = blockIdx.z;
    const int zb = z / NH_C;
    const int zh = z % NH_C;
    Ag += (long long)zb * sAb + (long long)zh * sAh;
    Bg += (long long)zb * sBb + (long long)zh * sBh;
    if (Cf) Cf += (long long)zb * sCb  + (long long)zh * sCh;
    if (Ch) Ch += (long long)zb * sChb + (long long)zh * sChh;

    const int wave = threadIdx.x >> 5;
    const int lane = threadIdx.x & 31;
    const int half = lane >> 4;      // 0: lanes 0-15, 1: lanes 16-31
    const int rm   = lane & 15;

    const int tm = blockIdx.y * 64  + (wave >> 2) * 32;   // wave row base (32 rows)
    const int tn = blockIdx.x * 128 + (wave & 3) * 32;    // wave col base (32 cols)
    if (tm >= M || tn >= N) return;                       // wave-uniform early-out

    const int m0 = tm + rm,      m1 = tm + 16 + rm;
    const int n0 = tn + rm,      n1 = tn + 16 + rm;
    const int m0l = (m0 < M) ? m0 : (M - 1);
    const int m1l = (m1 < M) ? m1 : (M - 1);
    const int n0l = (n0 < N) ? n0 : (N - 1);
    const int n1l = (n1 < N) ? n1 : (N - 1);

    const _Float16* __restrict__ arow0 = Ag + (size_t)m0l * lda;
    const _Float16* __restrict__ arow1 = Ag + (size_t)m1l * lda;
    const _Float16* __restrict__ brow0 = Bg + (size_t)n0l * ldb;
    const _Float16* __restrict__ brow1 = Bg + (size_t)n1l * ldb;

    v8f c00 = {}, c01 = {}, c10 = {}, c11 = {};
    for (int k = 0; k < Kd; k += 32) {
        // A fragment (16x32): lanes 0-15 hold K = k..k+7 and k+16..k+23 (row m)
        //                     lanes 16-31 hold K = k+8..k+15 and k+24..k+31
        const int ka = k + half * 8;
        // B fragment (32x16): lanes 0-15 hold K = k..k+7 and k+16..k+23 of column n,
        //                     lanes 16-31 hold K = k+8..k+15 and k+24..k+31 (Bg is B^T row-major)
        const int kb = k + half * 8;
        union { v16h v; v8h h[2]; } a0, a1, b0, b1;
        a0.h[0] = *(const v8h*)(arow0 + ka);
        a0.h[1] = *(const v8h*)(arow0 + ka + 16);
        a1.h[0] = *(const v8h*)(arow1 + ka);
        a1.h[1] = *(const v8h*)(arow1 + ka + 16);
        b0.h[0] = *(const v8h*)(brow0 + kb);
        b0.h[1] = *(const v8h*)(brow0 + kb + 16);
        b1.h[0] = *(const v8h*)(brow1 + kb);
        b1.h[1] = *(const v8h*)(brow1 + kb + 16);
        c00 = __builtin_amdgcn_wmma_f32_16x16x32_f16(false, a0.v, false, b0.v,
                                                     (short)0, c00, false, false);
        c01 = __builtin_amdgcn_wmma_f32_16x16x32_f16(false, a0.v, false, b1.v,
                                                     (short)0, c01, false, false);
        c10 = __builtin_amdgcn_wmma_f32_16x16x32_f16(false, a1.v, false, b0.v,
                                                     (short)0, c10, false, false);
        c11 = __builtin_amdgcn_wmma_f32_16x16x32_f16(false, a1.v, false, b1.v,
                                                     (short)0, c11, false, false);
    }

    // Epilogue. C layout: lanes 0-15 -> N=lane, VGPR r -> M=r; lanes 16-31 -> M=8+r.
    auto epi = [&](const v8f& cc, int tmS, int col) {
        const float bval = (bias && col < N) ? bias[col] : 0.0f;
#pragma unroll
        for (int r = 0; r < 8; ++r) {
            const int row = tmS + half * 8 + r;
            if (row >= M || col >= N) continue;
            float v = cc[r] * alpha + bval;
            if (residual) v += residual[(size_t)row * ldr + col];
            if (flags & F_GELU) v = 0.5f * v * (1.0f + erff(v * 0.70710678118f));
            if (Cf) Cf[(size_t)row * ldc + col] = v;
            if (Ch) {
                if (flags & F_STORE_VT) {
                    const int bb = row / T_C, tt = row % T_C;
                    const int hh = col / DH_C, dd = col % DH_C;
                    Ch[(((size_t)(bb * NH_C + hh) * DH_C + dd) * T_C) + tt] = (_Float16)v;
                } else {
                    Ch[(size_t)row * ldch + col] = (_Float16)v;
                }
            }
        }
    };
    epi(c00, tm,      tn + rm);
    epi(c01, tm,      tn + 16 + rm);
    epi(c10, tm + 16, tn + rm);
    epi(c11, tm + 16, tn + 16 + rm);
}

// ---------------- fp32 (KxN) -> f16 (NxK) transpose-convert ----------------
__global__ __launch_bounds__(256)
void transpose_cvt_kernel(const float* __restrict__ W, _Float16* __restrict__ WT,
                          int K, int N)
{
    __shared__ float tile[32][33];
    const int nBase = blockIdx.x * 32;
    const int kBase = blockIdx.y * 32;
#pragma unroll
    for (int j = 0; j < 4; ++j) {
        const int r = kBase + threadIdx.y + j * 8;
        const int c = nBase + threadIdx.x;
        tile[threadIdx.y + j * 8][threadIdx.x] =
            (r < K && c < N) ? W[(size_t)r * N + c] : 0.0f;
    }
    __syncthreads();
#pragma unroll
    for (int j = 0; j < 4; ++j) {
        const int r = nBase + threadIdx.y + j * 8;  // row of WT (n)
        const int c = kBase + threadIdx.x;          // col of WT (k)
        if (r < N && c < K)
            WT[(size_t)r * K + c] = (_Float16)tile[threadIdx.x][threadIdx.y + j * 8];
    }
}

// ---------------- embedding sum ----------------
__global__ __launch_bounds__(256)
void embed_kernel(const int* __restrict__ ids, const int* __restrict__ seg,
                  const float* __restrict__ we, const float* __restrict__ pe,
                  const float* __restrict__ te, float* __restrict__ y)
{
    const int row = blockIdx.x;          // 0..M_C-1, row = b*T + t
    const int t   = row % T_C;
    const int id  = ids[row];
    const int sg  = seg[row];
    const float* __restrict__ w = we + (size_t)id * H_C;
    const float* __restrict__ p = pe + (size_t)t  * H_C;
    const float* __restrict__ s = te + (size_t)sg * H_C;
    for (int i = threadIdx.x; i < H_C; i += 256)
        y[(size_t)row * H_C + i] = w[i] + p[i] + s[i];
}

// ---------------- LayerNorm (fp32 in -> fp32 + f16 out) ----------------
__global__ __launch_bounds__(256)
void ln_kernel(const float* __restrict__ y, const float* __restrict__ g,
               const float* __restrict__ b, float* __restrict__ x,
               _Float16* __restrict__ xh)
{
    __shared__ float red[256];
    const int row = blockIdx.x;
    const float* __restrict__ yr = y + (size_t)row * H_C;

    float s = 0.0f;
    for (int i = threadIdx.x; i < H_C; i += 256) s += yr[i];
    red[threadIdx.x] = s; __syncthreads();
    for (int o = 128; o > 0; o >>= 1) {
        if (threadIdx.x < o) red[threadIdx.x] += red[threadIdx.x + o];
        __syncthreads();
    }
    const float mean = red[0] * (1.0f / H_C);
    __syncthreads();

    float v = 0.0f;
    for (int i = threadIdx.x; i < H_C; i += 256) { float d = yr[i] - mean; v += d * d; }
    red[threadIdx.x] = v; __syncthreads();
    for (int o = 128; o > 0; o >>= 1) {
        if (threadIdx.x < o) red[threadIdx.x] += red[threadIdx.x + o];
        __syncthreads();
    }
    const float rstd = rsqrtf(red[0] * (1.0f / H_C) + 1e-12f);

    for (int i = threadIdx.x; i < H_C; i += 256) {
        const float o = (yr[i] - mean) * rstd * g[i] + b[i];
        x [(size_t)row * H_C + i] = o;
        xh[(size_t)row * H_C + i] = (_Float16)o;
    }
}

// ---------------- masked softmax over T, fp32 scores -> f16 probs ----------------
__global__ __launch_bounds__(256)
void softmax_kernel(const float* __restrict__ S, const int* __restrict__ mask,
                    _Float16* __restrict__ P)
{
    __shared__ float red[256];
    const size_t row = blockIdx.x;                // z*T + t, z = b*NH + h
    const int    b   = (int)(row / ((size_t)NH_C * T_C));
    const int    i   = threadIdx.x;               // 256 == T_C
    const float* __restrict__ sr = S + row * T_C;

    float v = sr[i] + (mask[b * T_C + i] ? 0.0f : -10000.0f);
    red[i] = v; __syncthreads();
    for (int o = 128; o > 0; o >>= 1) {
        if (i < o) red[i] = fmaxf(red[i], red[i + o]);
        __syncthreads();
    }
    const float mx = red[0]; __syncthreads();
    const float e = __expf(v - mx);
    red[i] = e; __syncthreads();
    for (int o = 128; o > 0; o >>= 1) {
        if (i < o) red[i] += red[i + o];
        __syncthreads();
    }
    P[row * T_C + i] = (_Float16)(e / red[0]);
}

// ---------------- misc ----------------
__global__ void zero_kernel(float* p) { if (threadIdx.x == 0 && blockIdx.x == 0) p[0] = 0.0f; }

// ---------------- CRF forward + gold score -> loss ----------------
__global__ __launch_bounds__(64)
void crf_forward_kernel(const float* __restrict__ logits, const int* __restrict__ mask,
                        const int* __restrict__ labels, const float* __restrict__ trans,
                        float* __restrict__ out_loss)
{
    __shared__ float tr[KC * KC];
    __shared__ float alpha[KC];
    __shared__ float nxt[KC];
    const int b = blockIdx.x;
    const int j = threadIdx.x;
    for (int i = threadIdx.x; i < KC * KC; i += blockDim.x) tr[i] = trans[i];
    __syncthreads();

    if (j < KC)
        alpha[j] = tr[START_C * KC + j] + logits[((size_t)b * T_C) * KC + j];
    __syncthreads();

    for (int t = 1; t < T_C; ++t) {
        const int mt = mask[b * T_C + t];
        if (j < KC) {
            float mx = -1e30f;
            for (int i = 0; i < KC; ++i) mx = fmaxf(mx, alpha[i] + tr[i * KC + j]);
            float s = 0.0f;
            for (int i = 0; i < KC; ++i) s += expf(alpha[i] + tr[i * KC + j] - mx);
            const float v = mx + logf(s) + logits[((size_t)b * T_C + t) * KC + j];
            nxt[j] = mt ? v : alpha[j];
        }
        __syncthreads();
        if (j < KC) alpha[j] = nxt[j];
        __syncthreads();
    }

    if (j == 0) {
        float mx = -1e30f;
        for (int i = 0; i < KC; ++i) mx = fmaxf(mx, alpha[i] + tr[i * KC + END_C]);
        float s = 0.0f;
        for (int i = 0; i < KC; ++i) s += expf(alpha[i] + tr[i * KC + END_C] - mx);
        const float fwd = mx + logf(s);

        const int tag0 = labels[b * T_C];
        float score = tr[START_C * KC + tag0] + logits[((size_t)b * T_C) * KC + tag0];
        int prev = tag0, len = 0;
        for (int t = 0; t < T_C; ++t) len += mask[b * T_C + t];
        for (int t = 1; t < T_C; ++t) {
            const int   tg = labels[b * T_C + t];
            const float m  = (float)mask[b * T_C + t];
            score += (tr[prev * KC + tg] + logits[((size_t)b * T_C + t) * KC + tg]) * m;
            prev = tg;
        }
        int lastIdx = len - 1;
        lastIdx = lastIdx < 0 ? 0 : (lastIdx > T_C - 1 ? T_C - 1 : lastIdx);
        const int lastTag = labels[b * T_C + lastIdx];
        score += tr[lastTag * KC + END_C];
        atomicAdd(out_loss, (fwd - score) * (1.0f / (float)B_C));
    }
}

// ---------------- Viterbi decode ----------------
__global__ __launch_bounds__(64)
void viterbi_kernel(const float* __restrict__ logits, const int* __restrict__ mask,
                    const float* __restrict__ trans, int* __restrict__ bp,
                    float* __restrict__ out_pred)
{
    __shared__ float tr[KC * KC];
    __shared__ float alpha[KC];
    __shared__ float nxt[KC];
    const int b = blockIdx.x;
    const int j = threadIdx.x;
    for (int i = threadIdx.x; i < KC * KC; i += blockDim.x) tr[i] = trans[i];
    __syncthreads();

    if (j < KC)
        alpha[j] = tr[START_C * KC + j] + logits[((size_t)b * T_C) * KC + j];
    __syncthreads();

    for (int t = 1; t < T_C; ++t) {
        const int mt = mask[b * T_C + t];
        if (j < KC) {
            float best = -1e30f; int bi = 0;
            for (int i = 0; i < KC; ++i) {
                const float v = alpha[i] + tr[i * KC + j];
                if (v > best) { best = v; bi = i; }
            }
            const float vm = best + logits[((size_t)b * T_C + t) * KC + j];
            nxt[j] = mt ? vm : alpha[j];
            bp[((size_t)(t - 1) * B_C + b) * KC + j] = mt ? bi : j;
        }
        __syncthreads();
        if (j < KC) alpha[j] = nxt[j];
        __syncthreads();
    }

    if (j == 0) {
        float best = -1e30f; int last = 0;
        for (int i = 0; i < KC; ++i) {
            const float v = alpha[i] + tr[i * KC + END_C];
            if (v > best) { best = v; last = i; }
        }
        int tag = last;
        out_pred[(size_t)b * T_C + (T_C - 1)] = (float)tag;
        for (int t = T_C - 1; t >= 1; --t) {
            tag = bp[((size_t)(t - 1) * B_C + b) * KC + tag];
            out_pred[(size_t)b * T_C + (t - 1)] = (float)tag;
        }
    }
}

// =====================================================================
extern "C" void kernel_launch(void* const* d_in, const int* in_sizes, int n_in,
                              void* d_out, int out_size, void* d_ws, size_t ws_size,
                              hipStream_t stream)
{
    (void)in_sizes; (void)n_in; (void)out_size; (void)ws_size;

    const int*   ids = (const int*)d_in[0];
    const int*   msk = (const int*)d_in[1];
    const int*   seg = (const int*)d_in[2];
    const int*   lbl = (const int*)d_in[3];
    const float* we  = (const float*)d_in[4];
    const float* pe  = (const float*)d_in[5];
    const float* te  = (const float*)d_in[6];
    const float* eg  = (const float*)d_in[7];
    const float* ebv = (const float*)d_in[8];
    const float* Wq  = (const float*)d_in[9];
    const float* bq  = (const float*)d_in[10];
    const float* Wk  = (const float*)d_in[11];
    const float* bk  = (const float*)d_in[12];
    const float* Wv  = (const float*)d_in[13];
    const float* bv  = (const float*)d_in[14];
    const float* Wo  = (const float*)d_in[15];
    const float* bo  = (const float*)d_in[16];
    const float* g1  = (const float*)d_in[17];
    const float* bg1 = (const float*)d_in[18];
    const float* W1  = (const float*)d_in[19];
    const float* b1  = (const float*)d_in[20];
    const float* W2  = (const float*)d_in[21];
    const float* b2  = (const float*)d_in[22];
    const float* g2  = (const float*)d_in[23];
    const float* bg2 = (const float*)d_in[24];
    const float* Wh  = (const float*)d_in[25];
    const float* bh  = (const float*)d_in[26];
    const float* tr  = (const float*)d_in[27];
    float* out = (float*)d_out;

    // ---- workspace carve (256B aligned) ----
    char* p = (char*)d_ws;
    auto carve = [&](size_t bytes) -> void* {
        void* r = (void*)p;
        p += (bytes + 255) & ~(size_t)255;
        return r;
    };
    float*    x    = (float*)   carve((size_t)M_C * H_C * 4);
    float*    y    = (float*)   carve((size_t)M_C * H_C * 4);
    _Float16* xh   = (_Float16*)carve((size_t)M_C * H_C * 2);
    _Float16* qh   = (_Float16*)carve((size_t)M_C * H_C * 2);
    _Float16* kh   = (_Float16*)carve((size_t)M_C * H_C * 2);
    _Float16* vT   = (_Float16*)carve((size_t)M_C * H_C * 2);
    _Float16* ctxh = (_Float16*)carve((size_t)M_C * H_C * 2);
    float*    S    = (float*)   carve((size_t)B_C * NH_C * T_C * T_C * 4);
    _Float16* Ph   = (_Float16*)carve((size_t)B_C * NH_C * T_C * T_C * 2);
    _Float16* ffh  = (_Float16*)carve((size_t)M_C * FF_C * 2);
    _Float16* WqT  = (_Float16*)carve((size_t)H_C * H_C * 2);
    _Float16* WkT  = (_Float16*)carve((size_t)H_C * H_C * 2);
    _Float16* WvT  = (_Float16*)carve((size_t)H_C * H_C * 2);
    _Float16* WoT  = (_Float16*)carve((size_t)H_C * H_C * 2);
    _Float16* W1T  = (_Float16*)carve((size_t)FF_C * H_C * 2);
    _Float16* W2T  = (_Float16*)carve((size_t)H_C * FF_C * 2);
    _Float16* WhT  = (_Float16*)carve((size_t)KC * H_C * 2);
    float*    lg   = (float*)   carve((size_t)M_C * KC * 4);
    int*      bp   = (int*)     carve((size_t)(T_C - 1) * B_C * KC * 4);

    auto gemm = [&](const _Float16* A, int lda, long long sAb, long long sAh,
                    const _Float16* B, int ldb, long long sBb, long long sBh,
                    const float* bias, const float* res, int ldr,
                    float* Cf, int ldc, long long sCb, long long sCh,
                    _Float16* Ch, int ldch, long long sChb, long long sChh,
                    int M, int N, int K, float alpha, int flags, int Z) {
        dim3 grid((N + 127) / 128, (M + 63) / 64, Z);
        wmma_gemm_kernel<<<grid, 256, 0, stream>>>(A, lda, sAb, sAh, B, ldb, sBb, sBh,
                                                   bias, res, ldr, Cf, ldc, sCb, sCh,
                                                   Ch, ldch, sChb, sChh, M, N, K, alpha, flags);
    };

    zero_kernel<<<1, 1, 0, stream>>>(out);
    embed_kernel<<<M_C, 256, 0, stream>>>(ids, seg, we, pe, te, y);
    ln_kernel<<<M_C, 256, 0, stream>>>(y, eg, ebv, x, xh);

    const dim3 tb(32, 8);
    const long long zAtt = (long long)T_C * H_C;       // per-batch stride of act rows
    const long long zS   = (long long)T_C * T_C;       // per-head stride in scores

    for (int l = 0; l < L_C; ++l) {
        const size_t wHH = (size_t)l * H_C * H_C;
        const size_t wHF = (size_t)l * H_C * FF_C;
        transpose_cvt_kernel<<<dim3(H_C / 32, H_C / 32), tb, 0, stream>>>(Wq + wHH, WqT, H_C, H_C);
        transpose_cvt_kernel<<<dim3(H_C / 32, H_C / 32), tb, 0, stream>>>(Wk + wHH, WkT, H_C, H_C);
        transpose_cvt_kernel<<<dim3(H_C / 32, H_C / 32), tb, 0, stream>>>(Wv + wHH, WvT, H_C, H_C);
        transpose_cvt_kernel<<<dim3(H_C / 32, H_C / 32), tb, 0, stream>>>(Wo + wHH, WoT, H_C, H_C);
        transpose_cvt_kernel<<<dim3(FF_C / 32, H_C / 32), tb, 0, stream>>>(W1 + wHF, W1T, H_C, FF_C);
        transpose_cvt_kernel<<<dim3(H_C / 32, FF_C / 32), tb, 0, stream>>>(W2 + wHF, W2T, FF_C, H_C);

        // Q, K, V projections (f16 outputs; V stored per-head transposed)
        gemm(xh, H_C, 0, 0, WqT, H_C, 0, 0, bq + (size_t)l * H_C, nullptr, 0,
             nullptr, 0, 0, 0, qh, H_C, 0, 0, M_C, H_C, H_C, 1.0f, 0, 1);
        gemm(xh, H_C, 0, 0, WkT, H_C, 0, 0, bk + (size_t)l * H_C, nullptr, 0,
             nullptr, 0, 0, 0, kh, H_C, 0, 0, M_C, H_C, H_C, 1.0f, 0, 1);
        gemm(xh, H_C, 0, 0, WvT, H_C, 0, 0, bv + (size_t)l * H_C, nullptr, 0,
             nullptr, 0, 0, 0, vT, 0, 0, 0, M_C, H_C, H_C, 1.0f, F_STORE_VT, 1);

        // scores = (q @ k^T) / sqrt(DH), batched over B*NH
        gemm(qh, H_C, zAtt, DH_C, kh, H_C, zAtt, DH_C, nullptr, nullptr, 0,
             S, T_C, (long long)NH_C * zS, zS, nullptr, 0, 0, 0,
             T_C, T_C, DH_C, 0.125f, 0, B_C * NH_C);

        softmax_kernel<<<B_C * NH_C * T_C, 256, 0, stream>>>(S, msk, Ph);

        // ctx = probs @ v, batched over B*NH; output back in (B*T, H) layout
        gemm(Ph, T_C, (long long)NH_C * zS, zS,
             vT, T_C, (long long)NH_C * DH_C * T_C, (long long)DH_C * T_C,
             nullptr, nullptr, 0, nullptr, 0, 0, 0,
             ctxh, H_C, zAtt, DH_C, T_C, DH_C, T_C, 1.0f, 0, B_C * NH_C);

        // O projection + residual -> y, then LN1 -> x/xh
        gemm(ctxh, H_C, 0, 0, WoT, H_C, 0, 0, bo + (size_t)l * H_C, x, H_C,
             y, H_C, 0, 0, nullptr, 0, 0, 0, M_C, H_C, H_C, 1.0f, 0, 1);
        ln_kernel<<<M_C, 256, 0, stream>>>(y, g1 + (size_t)l * H_C, bg1 + (size_t)l * H_C, x, xh);

        // FFN: GELU(x @ W1 + b1) @ W2 + b2 + residual, LN2
        gemm(xh, H_C, 0, 0, W1T, H_C, 0, 0, b1 + (size_t)l * FF_C, nullptr, 0,
             nullptr, 0, 0, 0, ffh, FF_C, 0, 0, M_C, FF_C, H_C, 1.0f, F_GELU, 1);
        gemm(ffh, FF_C, 0, 0, W2T, FF_C, 0, 0, b2 + (size_t)l * H_C, x, H_C,
             y, H_C, 0, 0, nullptr, 0, 0, 0, M_C, H_C, FF_C, 1.0f, 0, 1);
        ln_kernel<<<M_C, 256, 0, stream>>>(y, g2 + (size_t)l * H_C, bg2 + (size_t)l * H_C, x, xh);
    }

    // head logits = x @ W_head + b_head  (N = 35, handled by clamp/guards)
    transpose_cvt_kernel<<<dim3((KC + 31) / 32, H_C / 32), tb, 0, stream>>>(Wh, WhT, H_C, KC);
    gemm(xh, H_C, 0, 0, WhT, H_C, 0, 0, bh, nullptr, 0,
         lg, KC, 0, 0, nullptr, 0, 0, 0, M_C, KC, H_C, 1.0f, 0, 1);

    // CRF loss + Viterbi decode
    crf_forward_kernel<<<B_C, 64, 0, stream>>>(lg, msk, lbl, tr, out);
    viterbi_kernel<<<B_C, 64, 0, stream>>>(lg, msk, tr, bp, out + 1);
}
